// EncoderBlock_19318762897791
// MI455X (gfx1250) — compile-verified, matrix operand order checked by reading, not run on hardware
//
#include <hip/hip_runtime.h>
#include <hip/hip_bf16.h>

// ---------------------------------------------------------------------------
// Performer (FAVOR+) encoder block on gfx1250 (MI455X).
// All GEMMs run on v_wmma_f32_16x16x32_bf16 (bf16 operands, f32 accum).
// Big GEMMs stage block tiles through LDS (ds_store_b128 / ds_load_b128) with
// bank-conflict padding; global fragment traffic is minimal per block.
// ---------------------------------------------------------------------------

#define B_   32
#define L_   1024
#define HID_ 512
#define H_   8
#define DH_  64
#define FF_  2048
#define ROWS_ (B_ * L_)          // 32768
#define KEPS_ 1e-3f
#define LNEPS_ 1e-6f

#define SPAD_ 8                  // LDS row pad: 32 -> 40 bf16 (80B, 20 banks)
#define SLD_  (32 + SPAD_)

typedef unsigned short u16;
typedef __attribute__((ext_vector_type(16))) __bf16 v16bf;
typedef __attribute__((ext_vector_type(8)))  float  v8f;

union FragU { v16bf v; uint4 q[2]; };

__device__ __forceinline__ u16 bf16_from_f32(float f) {
  unsigned u = __float_as_uint(f);
  unsigned r = u + 0x7FFFu + ((u >> 16) & 1u);   // round-to-nearest-even
  return (u16)(r >> 16);
}
__device__ __forceinline__ float f32_from_bf16(u16 h) {
  return __uint_as_float(((unsigned)h) << 16);
}

// Load a 16x32 bf16 fragment from global (A layout; Bt rows identical).
// Lane (r, half): elements 0..7 = K half*8..+7, 8..15 = K 16+half*8..+7
__device__ __forceinline__ v16bf load_frag(const u16* base, int ld) {
  int lane = threadIdx.x & 31;
  int r    = lane & 15;
  int half = lane >> 4;
  const u16* p = base + (long)r * ld + half * 8;
  FragU f;
  f.q[0] = *(const uint4*)(p);
  f.q[1] = *(const uint4*)(p + 16);
  return f.v;
}

__device__ __forceinline__ v8f wmma_bf16(v16bf a, v16bf b, v8f c) {
  return __builtin_amdgcn_wmma_f32_16x16x32_bf16(
      /*neg_a=*/false, a, /*neg_b=*/false, b,
      /*c_mod=*/(short)0, c, /*reuse_a=*/false, /*reuse_b=*/false);
}

// ---------------------------------------------------------------------------
// Prep kernels
// ---------------------------------------------------------------------------
__global__ void convert_bf16_kernel(const float* __restrict__ in,
                                    u16* __restrict__ out, long n) {
  long t = (long)blockIdx.x * blockDim.x + threadIdx.x;
  if (t < n) out[t] = bf16_from_f32(in[t]);
}

// in f32 [R][C] row-major -> out bf16 [C][R]
__global__ void transpose_bf16_kernel(const float* __restrict__ in,
                                      u16* __restrict__ out, int R, int C) {
  long t = (long)blockIdx.x * blockDim.x + threadIdx.x;
  if (t >= (long)R * C) return;
  long r = t / C, c = t % C;
  out[c * (long)R + r] = bf16_from_f32(in[t]);
}

// ---------------------------------------------------------------------------
// Generic bf16 GEMM: C[row,col] = A[ROWS,K] x Bt[N,K]^T  (+bias, epilogue)
// Block: 256 threads = 8 waves (2x4), block tile 64x128, wave tile 32x32.
// A/B k-slabs staged through LDS once per k-step; next slab's global loads
// are issued before the WMMA work on the current slab (reg prefetch).
// MODE 0: bf16 row-major out    (qp, ff1)
// MODE 1: bf16 per-head transposed scatter -> [B,H,64,L]  (kpT, vT)
// MODE 2: f32 out with residual add (pre-LayerNorm buffers)
// ---------------------------------------------------------------------------
template <int MODE>
__global__ __launch_bounds__(256)
void gemm_kernel(const u16* __restrict__ A, int lda,
                 const u16* __restrict__ Bt, int ldb,
                 int K, int N,
                 const float* __restrict__ bias,
                 const float* __restrict__ resid,
                 int do_relu, float eps_add,
                 u16* __restrict__ out_b16,
                 float* __restrict__ out_f32) {
  __shared__ __align__(16) u16 sA[64][SLD_];    //  5.0 KB
  __shared__ __align__(16) u16 sB[128][SLD_];   // 10.0 KB

  int tid  = threadIdx.x;
  int wave = tid >> 5;
  int lane = tid & 31;
  int wrow = wave >> 2;                        // 0..1
  int wcol = wave & 3;                         // 0..3
  long row0 = (long)blockIdx.y * 64;
  long col0 = (long)blockIdx.x * 128;

  // Staging assignment: A 64x32 (16B/thread), B 128x32 (32B/thread).
  int arow = tid >> 2;                         // 0..63
  int acol = (tid & 3) * 8;                    // 0,8,16,24
  int brow = tid >> 1;                         // 0..127
  int bcol = (tid & 1) * 16;                   // 0,16
  const u16* Ag = A + (row0 + arow) * (long)lda + acol;
  const u16* Bg = Bt + (col0 + brow) * (long)ldb + bcol;

  uint4 ra  = *(const uint4*)(Ag);
  uint4 rb0 = *(const uint4*)(Bg);
  uint4 rb1 = *(const uint4*)(Bg + 8);

  int fr = lane & 15, fh = lane >> 4;          // fragment lane decode
  v8f acc[2][2] = {};
  int nk = K >> 5;
  for (int kt = 0; kt < nk; ++kt) {
    __syncthreads();                           // LDS buffer free
    *(uint4*)&sA[arow][acol]     = ra;
    *(uint4*)&sB[brow][bcol]     = rb0;
    *(uint4*)&sB[brow][bcol + 8] = rb1;
    __syncthreads();                           // tile visible
    if (kt + 1 < nk) {                         // prefetch next slab
      ra  = *(const uint4*)(Ag + (kt + 1) * 32);
      rb0 = *(const uint4*)(Bg + (kt + 1) * 32);
      rb1 = *(const uint4*)(Bg + (kt + 1) * 32 + 8);
    }
    FragU a0, a1, b0, b1;
    a0.q[0] = *(const uint4*)&sA[wrow * 32 + fr][fh * 8];
    a0.q[1] = *(const uint4*)&sA[wrow * 32 + fr][16 + fh * 8];
    a1.q[0] = *(const uint4*)&sA[wrow * 32 + 16 + fr][fh * 8];
    a1.q[1] = *(const uint4*)&sA[wrow * 32 + 16 + fr][16 + fh * 8];
    b0.q[0] = *(const uint4*)&sB[wcol * 32 + fr][fh * 8];
    b0.q[1] = *(const uint4*)&sB[wcol * 32 + fr][16 + fh * 8];
    b1.q[0] = *(const uint4*)&sB[wcol * 32 + 16 + fr][fh * 8];
    b1.q[1] = *(const uint4*)&sB[wcol * 32 + 16 + fr][16 + fh * 8];
    acc[0][0] = wmma_bf16(a0.v, b0.v, acc[0][0]);
    acc[0][1] = wmma_bf16(a0.v, b1.v, acc[0][1]);
    acc[1][0] = wmma_bf16(a1.v, b0.v, acc[1][0]);
    acc[1][1] = wmma_bf16(a1.v, b1.v, acc[1][1]);
  }

  int cn = lane & 15, half = lane >> 4;
#pragma unroll
  for (int i = 0; i < 2; ++i)
#pragma unroll
    for (int j = 0; j < 2; ++j)
#pragma unroll
      for (int r = 0; r < 8; ++r) {
        long row = row0 + wrow * 32 + i * 16 + half * 8 + r;
        long col = col0 + wcol * 32 + j * 16 + cn;
        float val = acc[i][j][r] + bias[col];
        if (do_relu) val = fmaxf(val, 0.0f);
        val += eps_add;
        if (MODE == 2) {
          val += resid[row * N + col];
          out_f32[row * N + col] = val;
        } else if (MODE == 0) {
          out_b16[row * N + col] = bf16_from_f32(val);
        } else {
          long h = col >> 6, m = col & 63;      // col = h*64+m
          long b = row >> 10, l = row & 1023;   // row = b*1024+l
          out_b16[(((b * H_ + h) * 64 + m) << 10) + l] = bf16_from_f32(val);
        }
      }
}

// ---------------------------------------------------------------------------
// kv[b,h] = kp^T (64xL) x v (Lx64); inputs kpT/vT are [B,H,64,L] bf16.
// Output stored transposed: kvT [B,H, d(64), m(64)] bf16.
// One workgroup per (b,h); 8 waves, each 16(m) x 32(d), K = L = 1024.
// ---------------------------------------------------------------------------
__global__ __launch_bounds__(256)
void kv_kernel(const u16* __restrict__ kpT, const u16* __restrict__ vT,
               u16* __restrict__ kvT) {
  int bh = blockIdx.x;                              // 0..255
  const u16* Ab = kpT + (long)bh * 64 * L_;
  const u16* Bb = vT + (long)bh * 64 * L_;
  int wave = threadIdx.x >> 5;
  int wr = wave >> 1;                               // m tile 0..3
  int wc = wave & 1;                                // d half 0..1
  const u16* A0 = Ab + (long)wr * 16 * L_;
  const u16* B0 = Bb + (long)wc * 32 * L_;
  v8f acc[2] = {};
  for (int k = 0; k < L_; k += 32) {
    v16bf a  = load_frag(A0 + k, L_);
    v16bf b0 = load_frag(B0 + k, L_);
    v16bf b1 = load_frag(B0 + (long)16 * L_ + k, L_);
    acc[0] = wmma_bf16(a, b0, acc[0]);
    acc[1] = wmma_bf16(a, b1, acc[1]);
  }
  int lane = threadIdx.x & 31;
  int cn = lane & 15, half = lane >> 4;
#pragma unroll
  for (int j = 0; j < 2; ++j)
#pragma unroll
    for (int r = 0; r < 8; ++r) {
      int m = wr * 16 + half * 8 + r;
      int d = wc * 32 + j * 16 + cn;
      kvT[((long)bh * 64 + d) * 64 + m] = bf16_from_f32(acc[j][r]);
    }
}

// ksum[b,h,m] = sum_l kpT[b,h,m,l]; one wave per row.
__global__ __launch_bounds__(256)
void ksum_kernel(const u16* __restrict__ kpT, float* __restrict__ ksum) {
  int row = blockIdx.x * 8 + (threadIdx.x >> 5);   // 0..16383
  int lane = threadIdx.x & 31;
  const u16* p = kpT + (long)row * L_;
  float s = 0.0f;
  for (int i = lane; i < L_; i += 32) s += f32_from_bf16(p[i]);
  for (int off = 16; off; off >>= 1) s += __shfl_xor(s, off, 32);
  if (lane == 0) ksum[row] = s;
}

// den[b,l,h] = dot(qp[b,l,h,:], ksum[b,h,:])
__global__ void den_kernel(const u16* __restrict__ qp,
                           const float* __restrict__ ksum,
                           float* __restrict__ den) {
  long t = (long)blockIdx.x * blockDim.x + threadIdx.x;  // (b*L+l)*H + h
  if (t >= (long)ROWS_ * H_) return;
  long h = t & (H_ - 1);
  long bl = t >> 3;
  long b = bl >> 10;
  const u16* q = qp + (bl * H_ + h) * 64;
  const float* ks = ksum + (b * H_ + h) * 64;
  float s = 0.0f;
#pragma unroll 8
  for (int m = 0; m < 64; ++m) s += f32_from_bf16(q[m]) * ks[m];
  den[t] = s;
}

// attn[b,l,h,d] = (qp(Lx64) x kv(64x64))[l,d] / den[b,l,h]
// Grid: (L/128, B*H); 8 waves, each 16(l) x 64(d), K = 64.
__global__ __launch_bounds__(256)
void attn_kernel(const u16* __restrict__ qp, const u16* __restrict__ kvT,
                 const float* __restrict__ den, u16* __restrict__ attn) {
  int bh = blockIdx.y;
  int b = bh >> 3, h = bh & 7;
  long l0 = (long)blockIdx.x * 128 + (threadIdx.x >> 5) * 16;
  const u16* A0 = qp + (((long)b * L_ + l0) * H_ + h) * 64;  // lda = 512
  const u16* B0 = kvT + (long)bh * 64 * 64;                  // ldb = 64
  v8f acc[4] = {};
  for (int k = 0; k < 64; k += 32) {
    v16bf a = load_frag(A0 + k, HID_);
#pragma unroll
    for (int j = 0; j < 4; ++j) {
      v16bf bf = load_frag(B0 + (long)j * 16 * 64 + k, 64);
      acc[j] = wmma_bf16(a, bf, acc[j]);
    }
  }
  int lane = threadIdx.x & 31;
  int cn = lane & 15, half = lane >> 4;
#pragma unroll
  for (int j = 0; j < 4; ++j)
#pragma unroll
    for (int r = 0; r < 8; ++r) {
      long l = l0 + half * 8 + r;
      int d = j * 16 + cn;
      float dv = den[((long)b * L_ + l) * H_ + h];
      attn[((long)b * L_ + l) * HID_ + h * 64 + d] =
          bf16_from_f32(acc[j][r] / dv);
    }
}

// LayerNorm over 512 cols; one wave per row; optional f32 and bf16 outputs.
__global__ __launch_bounds__(256)
void ln_kernel(const float* __restrict__ in, const float* __restrict__ scale,
               const float* __restrict__ bias, float* __restrict__ out_f32,
               u16* __restrict__ out_b16) {
  long row = (long)blockIdx.x * 8 + (threadIdx.x >> 5);
  int lane = threadIdx.x & 31;
  const float* p = in + row * HID_;
  float vals[16];
  float s = 0.0f;
#pragma unroll
  for (int i = 0; i < 16; ++i) { vals[i] = p[lane + i * 32]; s += vals[i]; }
  for (int off = 16; off; off >>= 1) s += __shfl_xor(s, off, 32);
  float mu = s * (1.0f / HID_);
  float v2 = 0.0f;
#pragma unroll
  for (int i = 0; i < 16; ++i) { float d = vals[i] - mu; v2 += d * d; }
  for (int off = 16; off; off >>= 1) v2 += __shfl_xor(v2, off, 32);
  float rstd = rsqrtf(v2 * (1.0f / HID_) + LNEPS_);
#pragma unroll
  for (int i = 0; i < 16; ++i) {
    int c = lane + i * 32;
    float o = (vals[i] - mu) * rstd * scale[c] + bias[c];
    if (out_f32) out_f32[row * HID_ + c] = o;
    if (out_b16) out_b16[row * HID_ + c] = bf16_from_f32(o);
  }
}

// ---------------------------------------------------------------------------
extern "C" void kernel_launch(void* const* d_in, const int* in_sizes, int n_in,
                              void* d_out, int out_size, void* d_ws,
                              size_t ws_size, hipStream_t stream) {
  (void)in_sizes; (void)n_in; (void)out_size; (void)ws_size;
  const float* x   = (const float*)d_in[0];
  const float* Wq  = (const float*)d_in[1];
  const float* bq  = (const float*)d_in[2];
  const float* Wk  = (const float*)d_in[3];
  const float* bk  = (const float*)d_in[4];
  const float* Wv  = (const float*)d_in[5];
  const float* bv  = (const float*)d_in[6];
  const float* Wo  = (const float*)d_in[7];
  const float* bo  = (const float*)d_in[8];
  const float* ln1s = (const float*)d_in[9];
  const float* ln1b = (const float*)d_in[10];
  const float* W1  = (const float*)d_in[11];
  const float* b1  = (const float*)d_in[12];
  const float* W2  = (const float*)d_in[13];
  const float* b2  = (const float*)d_in[14];
  const float* ln2s = (const float*)d_in[15];
  const float* ln2b = (const float*)d_in[16];
  float* out = (float*)d_out;

  char* w = (char*)d_ws;
  auto alloc = [&](size_t sz) {
    char* p = w;
    w += (sz + 255) & ~(size_t)255;
    return p;
  };
  u16*   xb   = (u16*)alloc((size_t)ROWS_ * HID_ * 2);
  u16*   wqT  = (u16*)alloc((size_t)HID_ * HID_ * 2);
  u16*   wkT  = (u16*)alloc((size_t)HID_ * HID_ * 2);
  u16*   wvT  = (u16*)alloc((size_t)HID_ * HID_ * 2);
  u16*   woT  = (u16*)alloc((size_t)HID_ * HID_ * 2);
  u16*   w1T  = (u16*)alloc((size_t)FF_ * HID_ * 2);
  u16*   w2T  = (u16*)alloc((size_t)HID_ * FF_ * 2);
  u16*   qp   = (u16*)alloc((size_t)ROWS_ * HID_ * 2);   // [B,L,H,M]
  u16*   kpT  = (u16*)alloc((size_t)ROWS_ * HID_ * 2);   // [B,H,M,L]
  u16*   vT   = (u16*)alloc((size_t)ROWS_ * HID_ * 2);   // [B,H,D,L]
  u16*   kvT  = (u16*)alloc((size_t)B_ * H_ * 64 * 64 * 2);
  float* ksum = (float*)alloc((size_t)B_ * H_ * 64 * 4);
  float* den  = (float*)alloc((size_t)ROWS_ * H_ * 4);
  u16*   attn = (u16*)alloc((size_t)ROWS_ * HID_ * 2);
  float* tmp  = (float*)alloc((size_t)ROWS_ * HID_ * 4); // pre-LN (reused)
  float* hf   = (float*)alloc((size_t)ROWS_ * HID_ * 4);
  u16*   hb   = (u16*)alloc((size_t)ROWS_ * HID_ * 2);
  u16*   ff1  = (u16*)alloc((size_t)ROWS_ * FF_ * 2);

  // ---- prep: bf16 conversions / transposes
  {
    long n = (long)ROWS_ * HID_;
    convert_bf16_kernel<<<(n + 255) / 256, 256, 0, stream>>>(x, xb, n);
    transpose_bf16_kernel<<<(512 * 512 + 255) / 256, 256, 0, stream>>>(Wq, wqT, 512, 512);
    transpose_bf16_kernel<<<(512 * 512 + 255) / 256, 256, 0, stream>>>(Wk, wkT, 512, 512);
    transpose_bf16_kernel<<<(512 * 512 + 255) / 256, 256, 0, stream>>>(Wv, wvT, 512, 512);
    transpose_bf16_kernel<<<(512 * 512 + 255) / 256, 256, 0, stream>>>(Wo, woT, 512, 512);
    transpose_bf16_kernel<<<(512 * 2048 + 255) / 256, 256, 0, stream>>>(W1, w1T, 512, 2048);
    transpose_bf16_kernel<<<(2048 * 512 + 255) / 256, 256, 0, stream>>>(W2, w2T, 2048, 512);
  }

  dim3 g512(HID_ / 128, ROWS_ / 64);   // N=512 GEMMs
  dim3 gff(FF_ / 128, ROWS_ / 64);     // N=2048 GEMM

  // ---- QKV projections (relu+eps on q/k; per-head transpose for k/v)
  gemm_kernel<0><<<g512, 256, 0, stream>>>(xb, HID_, wqT, HID_, HID_, HID_,
                                           bq, nullptr, 1, KEPS_, qp, nullptr);
  gemm_kernel<1><<<g512, 256, 0, stream>>>(xb, HID_, wkT, HID_, HID_, HID_,
                                           bk, nullptr, 1, KEPS_, kpT, nullptr);
  gemm_kernel<1><<<g512, 256, 0, stream>>>(xb, HID_, wvT, HID_, HID_, HID_,
                                           bv, nullptr, 0, 0.0f, vT, nullptr);

  // ---- FAVOR: kv, ksum, den, num/attn
  kv_kernel<<<B_ * H_, 256, 0, stream>>>(kpT, vT, kvT);
  ksum_kernel<<<(B_ * H_ * 64) / 8, 256, 0, stream>>>(kpT, ksum);
  den_kernel<<<((long)ROWS_ * H_ + 255) / 256, 256, 0, stream>>>(qp, ksum, den);
  attn_kernel<<<dim3(L_ / 128, B_ * H_), 256, 0, stream>>>(qp, kvT, den, attn);

  // ---- output projection + residual, LN1
  gemm_kernel<2><<<g512, 256, 0, stream>>>(attn, HID_, woT, HID_, HID_, HID_,
                                           bo, x, 0, 0.0f, nullptr, tmp);
  ln_kernel<<<ROWS_ / 8, 256, 0, stream>>>(tmp, ln1s, ln1b, hf, hb);

  // ---- FFN
  gemm_kernel<0><<<gff, 256, 0, stream>>>(hb, HID_, w1T, HID_, HID_, HID_,
                                          b1, nullptr, 1, 0.0f, ff1, nullptr);
  gemm_kernel<2><<<g512, 256, 0, stream>>>(ff1, FF_, w2T, FF_, FF_, HID_,
                                           b2, hf, 0, 0.0f, nullptr, tmp);
  ln_kernel<<<ROWS_ / 8, 256, 0, stream>>>(tmp, ln2s, ln2b, out, nullptr);
}
